// ResNet_88888643158616
// MI455X (gfx1250) — 8-bit matrix operand order repaired and checked by reading, compiled, not run on hardware
//
#include <hip/hip_runtime.h>
#include <stddef.h>


typedef int v8i __attribute__((ext_vector_type(8)));

#define HWPIX 3136
#define NPIX  100352
#define NTILE (NPIX / 16)
#define XQ_BYTES  (25690112)
#define ACT_BYTES (6422528)


__device__ __forceinline__ v8i load_a32(const signed char* p) {
  const int4* q = (const int4*)p;
  int4 lo = q[0], hi = q[1];
  v8i r;
  r[0] = lo.x; r[1] = lo.y; r[2] = lo.z; r[3] = lo.w;
  r[4] = hi.x; r[5] = hi.y; r[6] = hi.z; r[7] = hi.w;
  return r;
}

__device__ __forceinline__ v8i load_b64(const signed char* pix, int half) {
  int4 l = *(const int4*)(pix + half * 16);
  int4 h = *(const int4*)(pix + 32 + half * 16);
  v8i r;
  r[0] = l.x; r[1] = l.y; r[2] = l.z; r[3] = l.w;
  r[4] = h.x; r[5] = h.y; r[6] = h.z; r[7] = h.w;
  return r;
}

__device__ __forceinline__ void stage_weights(signed char* lds,
                                              const signed char* __restrict__ g,
                                              int bytes) {
  for (int i = threadIdx.x * 16; i < bytes; i += 256 * 16)
    *(int4*)(lds + i) = *(const int4*)(g + i);
  __syncthreads();
}

__device__ __forceinline__ void bn_relu_store8(v8i acc,
                                               const float* __restrict__ alpha,
                                               const float* __restrict__ off,
                                               int cobase,
                                               signed char* __restrict__ dst) {
  union { signed char c[8]; int2 v; } u;
#pragma unroll
  for (int v2 = 0; v2 < 8; ++v2) {
    int co = cobase + v2;
    float f = rintf((alpha[co] * (float)acc[v2] + off[co]) * 0.0009765625f);
    f = fminf(fmaxf(f, 0.0f), 127.0f);
    u.c[v2] = (signed char)(int)f;
  }
  *(int2*)dst = u.v;
}


__global__ __launch_bounds__(256) void quant_k(const float* __restrict__ x,
                                               signed char* __restrict__ xq) {
  size_t i = (size_t)blockIdx.x * 256 + threadIdx.x;
  int hw   = (int)(i % HWPIX);
  int c    = (int)((i / HWPIX) & 255);
  int nimg = (int)(i / ((size_t)HWPIX * 256));
  xq[((size_t)nimg * HWPIX + hw) * 256 + c] = (signed char)(int)x[i];
}

__device__ __forceinline__ void a_unpack(int i, int& tile, int& mm, int& half,
                                         int& kofs) {
  tile = i >> 10;
  int r = i & 1023, lane = r >> 5, q = r & 31;
  int v = q >> 2, b = q & 3;
  half = lane >> 4;
  mm = lane & 15;
  kofs = ((v >> 2) << 5) + (half << 4) + ((v & 3) << 2) + b;
}

__global__ __launch_bounds__(256) void prep_w1_k(const int* __restrict__ s,
                                                 const int* __restrict__ m,
                                                 signed char* __restrict__ wp) {
  int i = blockIdx.x * 256 + threadIdx.x;
  int tile, mm, half, kofs;
  a_unpack(i, tile, mm, half, kofs);
  int mtile = tile >> 2, ks = tile & 3;
  int idx = (mtile * 16 + mm) * 256 + ks * 64 + kofs;
  wp[i] = (signed char)(m[idx] * (1 << s[idx]));
}

__global__ __launch_bounds__(256) void prep_w2_k(const int* __restrict__ s,
                                                 const int* __restrict__ m,
                                                 signed char* __restrict__ wp) {
  int i = blockIdx.x * 256 + threadIdx.x;
  int tile, mm, half, kofs;
  a_unpack(i, tile, mm, half, kofs);
  int mtile = tile / 9, tap = tile % 9;
  int kh = tap / 3, kw = tap % 3;
  int co = mtile * 16 + mm, ci = kofs;
  int idx = ((co * 64 + ci) * 3 + kh) * 3 + kw;
  wp[i] = (signed char)(m[idx] * (1 << s[idx]));
}

__global__ __launch_bounds__(256) void prep_w3_k(const int* __restrict__ s,
                                                 const int* __restrict__ m,
                                                 signed char* __restrict__ wp) {
  int i = blockIdx.x * 256 + threadIdx.x;
  int tile, mm, half, kofs;
  a_unpack(i, tile, mm, half, kofs);
  int idx = (tile * 16 + mm) * 64 + kofs;
  wp[i] = (signed char)(m[idx] * (1 << s[idx]));
}

__global__ __launch_bounds__(128) void prep_off_k(
    const float* b1, const int* q1, const float* b2, const int* q2,
    const float* b3, const int* q3, float* off1, float* off2, float* off3) {
  int i = blockIdx.x * 128 + threadIdx.x;
  if (i < 64)       off1[i]       = b1[i]       * exp2f((float)q1[i]       + 10.0f);
  else if (i < 128) off2[i - 64]  = b2[i - 64]  * exp2f((float)q2[i - 64]  + 10.0f);
  else              off3[i - 128] = b3[i - 128] * exp2f((float)q3[i - 128] + 10.0f);
}


__global__ __launch_bounds__(256) void conv1_k(
    const signed char* __restrict__ xq, const signed char* __restrict__ wp,
    const float* __restrict__ alpha, const float* __restrict__ off,
    signed char* __restrict__ act) {
  __shared__ __align__(16) signed char lw[16384];
  stage_weights(lw, wp, 16384);

  const int lane = threadIdx.x & 31, wave = threadIdx.x >> 5;
  const int half = lane >> 4;
  const int p = (blockIdx.x * 8 + wave) * 16 + (lane & 15);
  const signed char* bbase = xq + (size_t)p * 256;
  const signed char* abase = lw + lane * 32;

  v8i B[4];
#pragma unroll
  for (int ks = 0; ks < 4; ++ks) B[ks] = load_b64(bbase + ks * 64, half);

  v8i acc[4] = {};
#pragma unroll
  for (int ks = 0; ks < 4; ++ks) {
#pragma unroll
    for (int mt = 0; mt < 4; ++mt) {
      v8i A = load_a32(abase + (mt * 4 + ks) * 1024);
      acc[mt] = __builtin_amdgcn_wmma_i32_16x16x64_iu8(true, A, true, B[ks],
                                                       acc[mt], false, false);
    }
  }
#pragma unroll
  for (int mt = 0; mt < 4; ++mt) {
    int cobase = mt * 16 + half * 8;
    bn_relu_store8(acc[mt], alpha, off, cobase,
                   act + (size_t)p * 64 + cobase);
  }
}

__global__ __launch_bounds__(256) void conv2_k(
    const signed char* __restrict__ act1, const signed char* __restrict__ wp,
    const float* __restrict__ alpha, const float* __restrict__ off,
    signed char* __restrict__ act2) {
  __shared__ __align__(16) signed char lw[36864];
  stage_weights(lw, wp, 36864);

  const int lane = threadIdx.x & 31, wave = threadIdx.x >> 5;
  const int half = lane >> 4;
  const int p = (blockIdx.x * 8 + wave) * 16 + (lane & 15);
  const int nimg = p / HWPIX, hw = p % HWPIX, h = hw / 56, w = hw % 56;
  const signed char* abase = lw + lane * 32;

  v8i Bcur = {};
  {
    int hh = h - 1, ww = w - 1;
    if (((unsigned)hh < 56u) && ((unsigned)ww < 56u))
      Bcur = load_b64(act1 + (size_t)(nimg * HWPIX + hh * 56 + ww) * 64, half);
  }

  v8i acc[4] = {};
#pragma unroll
  for (int tap = 0; tap < 9; ++tap) {
    v8i Bnext = {};
    if (tap < 8) {
      int t2 = tap + 1;
      int hh = h + t2 / 3 - 1, ww = w + t2 % 3 - 1;
      if (((unsigned)hh < 56u) && ((unsigned)ww < 56u))
        Bnext = load_b64(act1 + (size_t)(nimg * HWPIX + hh * 56 + ww) * 64,
                         half);
    }
#pragma unroll
    for (int mt = 0; mt < 4; ++mt) {
      v8i A = load_a32(abase + (mt * 9 + tap) * 1024);
      acc[mt] = __builtin_amdgcn_wmma_i32_16x16x64_iu8(true, A, true, Bcur,
                                                       acc[mt], false, false);
    }
    Bcur = Bnext;
  }
#pragma unroll
  for (int mt = 0; mt < 4; ++mt) {
    int cobase = mt * 16 + half * 8;
    bn_relu_store8(acc[mt], alpha, off, cobase,
                   act2 + (size_t)p * 64 + cobase);
  }
}

__global__ __launch_bounds__(256) void conv3_k(
    const signed char* __restrict__ act2, const signed char* __restrict__ wp,
    const float* __restrict__ alpha, const float* __restrict__ off,
    const signed char* __restrict__ xq, float* __restrict__ out) {
  __shared__ __align__(16) signed char lw[16384];
  stage_weights(lw, wp, 16384);

  const int lane = threadIdx.x & 31, wave = threadIdx.x >> 5;
  const int half = lane >> 4;
  const int p = (blockIdx.x * 8 + wave) * 16 + (lane & 15);
  const int nimg = p / HWPIX, hw = p % HWPIX;
  const signed char* abase = lw + lane * 32;

  v8i B = load_b64(act2 + (size_t)p * 64, half);
#pragma unroll 4
  for (int mt = 0; mt < 16; ++mt) {
    v8i A = load_a32(abase + mt * 1024);
    v8i acc = {};
    acc = __builtin_amdgcn_wmma_i32_16x16x64_iu8(true, A, true, B, acc,
                                                 false, false);
    int cobase = mt * 16 + half * 8;
    union { int2 v; signed char c[8]; } idu;
    idu.v = *(const int2*)(xq + (size_t)p * 256 + cobase);
    float* obase = out + ((size_t)nimg * 256 + cobase) * HWPIX + hw;
#pragma unroll
    for (int v2 = 0; v2 < 8; ++v2) {
      int co = cobase + v2;
      float f = rintf((alpha[co] * (float)acc[v2] + off[co]) * 0.0009765625f);
      f = fminf(fmaxf(f, -128.0f), 127.0f);
      f += (float)idu.c[v2];
      f = fminf(fmaxf(f, 0.0f), 127.0f);
      obase[(size_t)v2 * HWPIX] = f;
    }
  }
}


extern "C" void kernel_launch(void* const* d_in, const int* in_sizes, int n_in,
                              void* d_out, int out_size, void* d_ws,
                              size_t ws_size, hipStream_t stream) {
  const float* x  = (const float*)d_in[0];
  const int*   s1 = (const int*)d_in[1];
  const int*   m1 = (const int*)d_in[2];
  const float* a1 = (const float*)d_in[3];
  const float* b1 = (const float*)d_in[4];
  const int*   q1 = (const int*)d_in[5];
  const int*   s2 = (const int*)d_in[6];
  const int*   m2 = (const int*)d_in[7];
  const float* a2 = (const float*)d_in[8];
  const float* b2 = (const float*)d_in[9];
  const int*   q2 = (const int*)d_in[10];
  const int*   s3 = (const int*)d_in[11];
  const int*   m3 = (const int*)d_in[12];
  const float* a3 = (const float*)d_in[13];
  const float* b3 = (const float*)d_in[14];
  const int*   q3 = (const int*)d_in[15];

  char* ws = (char*)d_ws;
  signed char* xq   = (signed char*)ws;
  signed char* act1 = (signed char*)(ws + XQ_BYTES);
  signed char* act2 = act1 + ACT_BYTES;
  signed char* wp1  = act2 + ACT_BYTES;
  signed char* wp2  = wp1 + 16384;
  signed char* wp3  = wp2 + 36864;
  float* off1 = (float*)(wp3 + 16384);
  float* off2 = off1 + 64;
  float* off3 = off2 + 64;

  prep_w1_k<<<64, 256, 0, stream>>>(s1, m1, wp1);
  prep_w2_k<<<144, 256, 0, stream>>>(s2, m2, wp2);
  prep_w3_k<<<64, 256, 0, stream>>>(s3, m3, wp3);
  prep_off_k<<<3, 128, 0, stream>>>(b1, q1, b2, q2, b3, q3, off1, off2, off3);
  quant_k<<<XQ_BYTES / 256, 256, 0, stream>>>(x, xq);

  conv1_k<<<NTILE / 8, 256, 0, stream>>>(xq, wp1, a1, off1, act1);
  conv2_k<<<NTILE / 8, 256, 0, stream>>>(act1, wp2, a2, off2, act2);
  conv3_k<<<NTILE / 8, 256, 0, stream>>>(act2, wp3, a3, off3, xq,
                                         (float*)d_out);
}
